// LocalEncoder_48558900249066
// MI455X (gfx1250) — matrix operand order repaired and checked by reading, not compiled, not run on hardware
//
#include <hip/hip_runtime.h>
#include <hip/hip_bf16.h>

typedef __attribute__((ext_vector_type(16))) __bf16 v16bf;
typedef __attribute__((ext_vector_type(8)))  __bf16 v8bf;
typedef __attribute__((ext_vector_type(8)))  float  v8f;

static __device__ __forceinline__ __bf16 f2bf(float f) {     // RNE (prep paths)
    unsigned u = __builtin_bit_cast(unsigned, f);
    u += 0x7FFFu + ((u >> 16) & 1u);
    unsigned short s = (unsigned short)(u >> 16);
    return __builtin_bit_cast(__bf16, s);
}

static __device__ __forceinline__ void store_bf16_trunc(__bf16* p, float f) {
    // truncating f32->bf16 store: lowers to global_store_d16_hi_b16, zero ALU
    unsigned u = __builtin_bit_cast(unsigned, f);
    *(unsigned short*)p = (unsigned short)(u >> 16);
}

static __device__ __forceinline__ v8f wmma_bf16(v16bf a, v16bf b, v8f c) {
    return __builtin_amdgcn_wmma_f32_16x16x32_bf16(false, a, false, b, (short)0, c, false, false);
}

// A fragment (16x32) from bf16 row-major [ld]: two contiguous 16B chunks per lane.
static __device__ __forceinline__ v16bf load_a_bf16(const __bf16* __restrict__ src, int ld,
                                                    int rowBase, int kBase, int lane, int maxRowIdx) {
    int half = lane >> 4;
    int row  = rowBase + (lane & 15);
    if (row > maxRowIdx) row = maxRowIdx;
    const __bf16* p = src + (unsigned)row * ld + kBase + half * 8;
    v8bf lo = *(const v8bf*)(p);
    v8bf hi = *(const v8bf*)(p + 16);
    return __builtin_shufflevector(lo, hi, 0,1,2,3,4,5,6,7, 8,9,10,11,12,13,14,15);
}

// A fragment from f32 source (only for aggr, which must stay f32 for atomics)
static __device__ __forceinline__ v16bf load_a_f32(const float* __restrict__ src, int ld,
                                                   int rowBase, int kBase, int lane, int maxRowIdx) {
    int half = lane >> 4;
    int row  = rowBase + (lane & 15);
    if (row > maxRowIdx) row = maxRowIdx;
    const float* p = src + (unsigned)row * ld + kBase + half * 8;
    v16bf a;
#pragma unroll
    for (int i = 0; i < 8; ++i) a[i] = f2bf(p[i]);
#pragma unroll
    for (int i = 0; i < 8; ++i) a[8 + i] = f2bf(p[16 + i]);
    return a;
}

// B fragment from pre-packed layout: packed[frag][lane][16] -> one 32B vector load per lane
static __device__ __forceinline__ v16bf load_b_packed(const __bf16* __restrict__ base, int frag, int lane) {
    return *(const v16bf*)(base + (unsigned)frag * 512 + lane * 16);
}

// ---------------- prep: pack node weights into fragment order; fused W2 = edge_W @ E_W; b2 ----------------
__global__ void prep_kernel(const float* __restrict__ embW, const float* __restrict__ UW,
                            const float* __restrict__ VW,   const float* __restrict__ AW,
                            const float* __restrict__ BW,   const float* __restrict__ edgeW,
                            const float* __restrict__ EW,   const float* __restrict__ edge_b,
                            const float* __restrict__ E_b,
                            __bf16* __restrict__ wp, __bf16* __restrict__ W2p, float* __restrict__ b2) {
    int tid = threadIdx.x, bid = blockIdx.x;
    if (bid < 40) {
        int base = (bid * 256 + tid) * 8;
#pragma unroll
        for (int t = 0; t < 8; ++t) {
            int flat = base + t;
            if (flat < 5 * 16384) {
                int m    = flat >> 14;
                int kf   = (flat >> 12) & 3;
                int nt   = (flat >> 9) & 7;
                int lane = (flat >> 4) & 31;
                int j    = flat & 15;
                int k    = kf * 32 + (j >> 3) * 16 + (lane >> 4) * 8 + (j & 7);
                int n    = nt * 16 + (lane & 15);
                const float* src = (m == 0) ? embW : (m == 1) ? UW : (m == 2) ? VW
                                 : (m == 3) ? AW : BW;
                wp[flat] = f2bf(src[k * 128 + n]);
            }
        }
    } else {
        for (int o = tid; o < 4096; o += 256) {
            int j = o & 15, lane = (o >> 4) & 31, nt = o >> 9;
            int k = (j >> 3) * 16 + (lane >> 4) * 8 + (j & 7);
            int n = nt * 16 + (lane & 15);
            float s = 0.f;
            if (k < 16)
                for (int kk = 0; kk < 128; ++kk) s += edgeW[k * 128 + kk] * EW[kk * 128 + n];
            W2p[o] = f2bf(s);
        }
        if (tid < 128) {
            float s = E_b[tid];
            for (int k = 0; k < 128; ++k) s += edge_b[k] * EW[k * 128 + tid];
            b2[tid] = s;
        }
    }
}

__global__ void f32_to_bf16_kernel(const float* __restrict__ src, __bf16* __restrict__ dst, int n) {
    int i0 = (blockIdx.x * 256 + threadIdx.x) * 8;
#pragma unroll
    for (int j = 0; j < 8; ++j) {
        int i = i0 + j;
        if (i < n) dst[i] = f2bf(src[i]);
    }
}

__global__ void zero_kernel(float* __restrict__ p, int n) {
    int i = blockIdx.x * blockDim.x + threadIdx.x;
    if (i < n) p[i] = 0.f;
}

// ---------------- node kernel: h/hb = x@embW+b; Vh/Ah/Uh = h@{V,A,U}+b ----------------
__global__ void node_kernel(const __bf16* __restrict__ xb, const __bf16* __restrict__ wp,
                            const float* __restrict__ emb_b, const float* __restrict__ U_b,
                            const float* __restrict__ V_b,   const float* __restrict__ A_b,
                            float* __restrict__ h, __bf16* __restrict__ hb,
                            float* __restrict__ Vh, float* __restrict__ Ah,
                            float* __restrict__ Uh, int N) {
    int tid  = threadIdx.x;
    int wave = __builtin_amdgcn_readfirstlane(tid >> 5);   // scalar wave id
    int lane = tid & 31;
    int rowBase = (blockIdx.x * 8 + wave) * 16;            // scalar
    if (rowBase >= N) return;                              // scalar branch
    int half = lane >> 4, nlo = lane & 15, maxRow = N - 1;
    bool full = (rowBase + 16 <= N);                       // scalar
    unsigned rbase = (unsigned)(rowBase + half * 8) * 128 + nlo;

    v16bf a[4];
#pragma unroll
    for (int kf = 0; kf < 4; ++kf) a[kf] = load_a_bf16(xb, 128, rowBase, kf * 32, lane, maxRow);

    // Stage 1: h (f32) + hb (bf16, truncated store)
#pragma unroll
    for (int nt = 0; nt < 8; ++nt) {
        v8f c = {};
#pragma unroll
        for (int kf = 0; kf < 4; ++kf)
            c = wmma_bf16(a[kf], load_b_packed(wp, kf * 8 + nt, lane), c);
        int col = nt * 16 + nlo;
        float bias = emb_b[col];
        float*  hp  = h  + rbase + nt * 16;
        __bf16* hbp = hb + rbase + nt * 16;
        if (full) {
#pragma unroll
            for (int r = 0; r < 8; ++r) {
                float v = c[r] + bias;
                hp[r * 128u] = v;
                store_bf16_trunc(hbp + r * 128u, v);
            }
        } else {
#pragma unroll
            for (int r = 0; r < 8; ++r) {
                int row = rowBase + r + half * 8;
                if (row < N) {
                    float v = c[r] + bias;
                    hp[r * 128u] = v;
                    store_bf16_trunc(hbp + r * 128u, v);
                }
            }
        }
    }

    // Stage 2: reload this wave's rows from hb (same-wave store->load, ordered)
#pragma unroll
    for (int kf = 0; kf < 4; ++kf) a[kf] = load_a_bf16(hb, 128, rowBase, kf * 32, lane, maxRow);

    const int    mf[3]   = { 2, 3, 1 };              // V, A, U matrix slots in wp
    const float* bs[3]   = { V_b, A_b, U_b };
    float*       out3[3] = { Vh, Ah, Uh };
#pragma unroll
    for (int m = 0; m < 3; ++m) {
        const int fbase = mf[m] * 32;
#pragma unroll
        for (int nt = 0; nt < 8; ++nt) {
            v8f c = {};
#pragma unroll
            for (int kf = 0; kf < 4; ++kf)
                c = wmma_bf16(a[kf], load_b_packed(wp, fbase + kf * 8 + nt, lane), c);
            int col = nt * 16 + nlo;
            float bias = bs[m][col];
            float* op = out3[m] + rbase + nt * 16;
            if (full) {
#pragma unroll
                for (int r = 0; r < 8; ++r) op[r * 128u] = c[r] + bias;
            } else {
#pragma unroll
                for (int r = 0; r < 8; ++r) {
                    int row = rowBase + r + half * 8;
                    if (row < N) op[r * 128u] = c[r] + bias;
                }
            }
        }
    }
}

// ---------------- edge kernel: preact via WMMA (attr@W2), gather, sigmoid, scatter-add ----------------
__global__ void edge_kernel(const float* __restrict__ edge_attr, const int* __restrict__ ei,
                            const __bf16* __restrict__ W2p, const float* __restrict__ b2,
                            const float* __restrict__ Vh, const float* __restrict__ Ah,
                            const float* __restrict__ Uh, float* __restrict__ aggr, int E) {
    int tid  = threadIdx.x;
    int wave = __builtin_amdgcn_readfirstlane(tid >> 5);   // scalar wave id
    int lane = tid & 31;
    int half = lane >> 4, nlo = lane & 15;
    int tileBase = (blockIdx.x * 8 + wave) * 16;           // scalar
    if (tileBase >= E) return;                             // scalar branch
    bool full = (tileBase + 16 <= E);                      // scalar

    // A fragment: 16 edges x K=16 (padded to 32 with zeros)
    int arow = tileBase + nlo; if (arow > E - 1) arow = E - 1;
    const float* p = edge_attr + (unsigned)arow * 16 + half * 8;
    v16bf af;
#pragma unroll
    for (int i = 0; i < 8; ++i) af[i] = f2bf(p[i]);
#pragma unroll
    for (int i = 0; i < 8; ++i) af[8 + i] = (__bf16)0.0f;

    // all 8 accumulators up-front (one WMMA per 16-channel group)
    v8f c[8];
#pragma unroll
    for (int nt = 0; nt < 8; ++nt)
        c[nt] = wmma_bf16(af, load_b_packed(W2p, nt, lane), (v8f){});

    float b2v[8];
#pragma unroll
    for (int nt = 0; nt < 8; ++nt) b2v[nt] = b2[nt * 16 + nlo];

    int dst8[8], src8[8];
#pragma unroll
    for (int r = 0; r < 8; ++r) {
        int e = tileBase + r + half * 8; if (e > E - 1) e = E - 1;
        src8[r] = ei[e];       // edge_index[0][e]
        dst8[r] = ei[E + e];   // edge_index[1][e]
    }

    // r outer: fixed src/dst -> base pointers, all gathers at constant offsets nt*64B
#pragma unroll
    for (int r = 0; r < 8; ++r) {
        if (!full) { int e = tileBase + r + half * 8; if (e >= E) continue; }
        unsigned di = (unsigned)dst8[r] * 128 + nlo;
        unsigned si = (unsigned)src8[r] * 128 + nlo;
        const float* pV = Vh   + di;
        const float* pA = Ah   + si;
        const float* pU = Uh   + si;
        float*       pG = aggr + di;
#pragma unroll
        for (int nt = 0; nt < 8; ++nt) {
            float pre = c[nt][r] + b2v[nt] + pV[nt * 16] + pA[nt * 16];
            float g = 1.0f / (1.0f + __expf(-pre));
            atomicAdd(&pG[nt * 16], pU[nt * 16] * g);
        }
    }
}

// ---------------- output projection: tmp = h + aggr@B_W + B_b ----------------
__global__ void outproj_kernel(const float* __restrict__ aggr, const __bf16* __restrict__ wp,
                               const float* __restrict__ B_b, const float* __restrict__ h,
                               float* __restrict__ tmp, int N) {
    int tid  = threadIdx.x;
    int wave = __builtin_amdgcn_readfirstlane(tid >> 5);
    int lane = tid & 31;
    int rowBase = (blockIdx.x * 8 + wave) * 16;
    if (rowBase >= N) return;
    int half = lane >> 4, nlo = lane & 15, maxRow = N - 1;
    bool full = (rowBase + 16 <= N);
    unsigned rbase = (unsigned)(rowBase + half * 8) * 128 + nlo;

    v16bf a[4];
#pragma unroll
    for (int kf = 0; kf < 4; ++kf) a[kf] = load_a_f32(aggr, 128, rowBase, kf * 32, lane, maxRow);

    const int fbase = 4 * 32;   // B_W slot
#pragma unroll
    for (int nt = 0; nt < 8; ++nt) {
        v8f c = {};
#pragma unroll
        for (int kf = 0; kf < 4; ++kf)
            c = wmma_bf16(a[kf], load_b_packed(wp, fbase + kf * 8 + nt, lane), c);
        int col = nt * 16 + nlo;
        float bias = B_b[col];
        const float* hp = h   + rbase + nt * 16;
        float*       tp = tmp + rbase + nt * 16;
        if (full) {
#pragma unroll
            for (int r = 0; r < 8; ++r) tp[r * 128u] = c[r] + bias + hp[r * 128u];
        } else {
#pragma unroll
            for (int r = 0; r < 8; ++r) {
                int row = rowBase + r + half * 8;
                if (row < N) tp[r * 128u] = c[r] + bias + hp[r * 128u];
            }
        }
    }
}

// ---------------- LayerNorm: one wave32 per row ----------------
__global__ void ln_kernel(const float* __restrict__ tmp, const float* __restrict__ g,
                          const float* __restrict__ b, float* __restrict__ out, int N) {
    int wave = __builtin_amdgcn_readfirstlane(threadIdx.x >> 5);
    int lane = threadIdx.x & 31;
    int row = blockIdx.x * 8 + wave;                       // scalar
    if (row >= N) return;                                  // scalar branch
    const float* p = tmp + (unsigned)row * 128;
    float v[4], s1 = 0.f, s2 = 0.f;
#pragma unroll
    for (int i = 0; i < 4; ++i) { v[i] = p[lane + 32 * i]; s1 += v[i]; s2 += v[i] * v[i]; }
#pragma unroll
    for (int off = 16; off > 0; off >>= 1) {
        s1 += __shfl_xor(s1, off, 32);
        s2 += __shfl_xor(s2, off, 32);
    }
    float mu  = s1 * (1.0f / 128.0f);
    float var = s2 * (1.0f / 128.0f) - mu * mu;
    float rs  = rsqrtf(var + 1e-5f);
    float* o = out + (unsigned)row * 128;
#pragma unroll
    for (int i = 0; i < 4; ++i) {
        int c = lane + 32 * i;
        o[c] = (v[i] - mu) * rs * g[c] + b[c];
    }
}

extern "C" void kernel_launch(void* const* d_in, const int* in_sizes, int n_in,
                              void* d_out, int out_size, void* d_ws, size_t ws_size,
                              hipStream_t stream) {
    const float* x         = (const float*)d_in[0];
    const int*   edge_index= (const int*)  d_in[1];
    const float* edge_attr = (const float*)d_in[2];
    const float* emb_W     = (const float*)d_in[3];
    const float* emb_b     = (const float*)d_in[4];
    const float* edge_W    = (const float*)d_in[5];
    const float* edge_b    = (const float*)d_in[6];
    const float* U_W       = (const float*)d_in[7];
    const float* U_b       = (const float*)d_in[8];
    const float* V_W       = (const float*)d_in[9];
    const float* V_b       = (const float*)d_in[10];
    const float* A_W       = (const float*)d_in[11];
    const float* A_b       = (const float*)d_in[12];
    const float* B_W       = (const float*)d_in[13];
    const float* B_b       = (const float*)d_in[14];
    const float* E_W       = (const float*)d_in[15];
    const float* E_b       = (const float*)d_in[16];
    const float* ln_g      = (const float*)d_in[17];
    const float* ln_b      = (const float*)d_in[18];
    float* out = (float*)d_out;

    int N = in_sizes[0] / 128;
    int E = in_sizes[2] / 16;
    size_t NH = (size_t)N * 128;

    float* wsf  = (float*)d_ws;
    float* h    = wsf;
    float* Vh   = wsf + NH;
    float* Ah   = wsf + 2 * NH;
    float* Uh   = wsf + 3 * NH;
    float* aggr = wsf + 4 * NH;
    __bf16* wp  = (__bf16*)(wsf + 5 * NH);   // packed fragments: emb,U,V,A,B (5 * 16384)
    __bf16* W2p = wp + 5 * 16384;            // packed W2 fragments (4096)
    float*  b2  = (float*)(W2p + 4096);
    __bf16* xb  = (__bf16*)(b2 + 128);       // bf16 copy of x (N*128)
    __bf16* hb  = xb + NH;                   // bf16 copy of h (N*128)
    float*  tmp = Vh;                        // reuse (Vh dead after edge kernel)

    prep_kernel<<<41, 256, 0, stream>>>(emb_W, U_W, V_W, A_W, B_W, edge_W, E_W,
                                        edge_b, E_b, wp, W2p, b2);
    int totalNH = N * 128;
    f32_to_bf16_kernel<<<(totalNH / 8 + 255) / 256, 256, 0, stream>>>(x, xb, totalNH);
    zero_kernel<<<(totalNH + 255) / 256, 256, 0, stream>>>(aggr, totalNH);

    int nodeBlocks = (N + 127) / 128;
    node_kernel<<<nodeBlocks, 256, 0, stream>>>(xb, wp, emb_b, U_b, V_b, A_b,
                                                h, hb, Vh, Ah, Uh, N);

    int edgeBlocks = (E + 127) / 128;
    edge_kernel<<<edgeBlocks, 256, 0, stream>>>(edge_attr, edge_index, W2p, b2,
                                                Vh, Ah, Uh, aggr, E);

    outproj_kernel<<<nodeBlocks, 256, 0, stream>>>(aggr, wp, B_b, h, tmp, N);

    ln_kernel<<<(N + 7) / 8, 256, 0, stream>>>(tmp, ln_g, ln_b, out, N);
}
